// set_attention_20212116095168
// MI455X (gfx1250) — compile-verified, matrix operand order checked by reading, not run on hardware
//
#include <hip/hip_runtime.h>
#include <math.h>

// ---------------------------------------------------------------------------
// Set-attention for MI455X (gfx1250, wave32, WMMA).
// Pipeline: Q projection + fused K/V projection (f32->f16, WMMA f32 accum)
// -> fused flash attention with exact-zero masking -> output projection.
// All matrix math uses v_wmma_f32_16x16x32_f16; K-tile staging uses
// GLOBAL_LOAD_ASYNC_TO_LDS_B128. Fragment gathers are indexed in dword
// units with constant per-register offsets so they merge into b128 loads.
// ---------------------------------------------------------------------------

typedef __attribute__((ext_vector_type(16))) _Float16 v16h;
typedef __attribute__((ext_vector_type(8)))  float    v8f;
typedef __attribute__((ext_vector_type(4)))  int      v4i;

union AF { v16h v; unsigned u[8]; };

#if __has_builtin(__builtin_amdgcn_global_load_async_to_lds_b128) && \
    __has_builtin(__builtin_amdgcn_s_wait_asynccnt)
#define USE_ASYNC_LDS 1
typedef __attribute__((address_space(1))) v4i* g128_t;  // global int4*
typedef __attribute__((address_space(3))) v4i* l128_t;  // LDS int4*
#endif

// Dword offset of fragment register j (0..7) relative to a base that already
// includes row*(stride/2) + ((lane>>4)<<2): per ISA 16-bit A/B layout,
// VGPR0..3 -> dwords 0..3, VGPR4..7 -> dwords 8..11 (lanes>=16 add 4 dwords,
// folded into the base). Constants {0,1,2,3,8,9,10,11} -> two b128 runs.
__device__ __forceinline__ int jd(int j) { return ((j & 4) << 1) | (j & 3); }

// LDS row strides in halves (PSTRIDE/2, ASTRIDE/2 dwords per row):
// multiples of 8 halves (16 B) keep b128 alignment provable; 28 / 36 dwords
// per row are bijective mod 64 banks -> conflict-free fragment gathers.
#define PSTRIDE 56   // projection tiles, 32-wide K panels (28 dwords/row)
#define ASTRIDE 72   // attention tiles, 64-wide panels  (36 dwords/row)

// ---------------------------------------------------------------------------
// Q projection: Q[b] = xf[b] @ W1  (512 x 512) x (512 x 512) -> f16.
// Workgroup tile 128(M) x 64(N); 8 waves, each 16 rows x 4 column tiles.
// batch b = sy*4+sx maps to x[sx][sy].
// ---------------------------------------------------------------------------
__global__ __launch_bounds__(256) void projq_kernel(
    const float* __restrict__ X, const float* __restrict__ W,
    _Float16* __restrict__ P)
{
    const int D = 512, M = 512;
    const int tm = blockIdx.x, tn = blockIdx.y, b = blockIdx.z;
    const int tid = threadIdx.x, lane = tid & 31, wave = tid >> 5;

    const int sx = b & 3, sy = b >> 2;
    const float* Xb = X + (size_t)(sx * 4 + sy) * M * D;
    _Float16*    Pb = P + (size_t)b * M * D;

    __shared__ alignas(16) _Float16 As[128 * PSTRIDE];  // [m][k]
    __shared__ alignas(16) _Float16 Bs[64 * PSTRIDE];   // [n][k] (W transposed)

    const int row0 = tm * 128, col0 = tn * 64;
    const int hb = (lane >> 4) << 2;   // +4 dwords for lanes 16..31

    v8f acc[4];
    for (int t = 0; t < 4; ++t)
        for (int v = 0; v < 8; ++v) acc[t][v] = 0.0f;

    const unsigned* As32 = (const unsigned*)As;
    const unsigned* Bs32 = (const unsigned*)Bs;

    for (int kb = 0; kb < D; kb += 32) {
        for (int i = tid; i < 128 * 32; i += 256) {
            int r = i >> 5, c = i & 31;
            As[r * PSTRIDE + c] = (_Float16)Xb[(size_t)(row0 + r) * D + kb + c];
        }
        for (int i = tid; i < 64 * 32; i += 256) {
            int n = i >> 5, c = i & 31;
            Bs[n * PSTRIDE + c] = (_Float16)W[(size_t)(kb + c) * D + col0 + n];
        }
        __syncthreads();

        AF a;
        const unsigned* ap = As32 + (wave * 16 + (lane & 15)) * (PSTRIDE / 2) + hb;
        for (int j = 0; j < 8; ++j) a.u[j] = ap[jd(j)];

        for (int t = 0; t < 4; ++t) {
            AF bf;
            const unsigned* bp = Bs32 + (t * 16 + (lane & 15)) * (PSTRIDE / 2) + hb;
            for (int j = 0; j < 8; ++j) bf.u[j] = bp[jd(j)];
            acc[t] = __builtin_amdgcn_wmma_f32_16x16x32_f16(
                false, a.v, false, bf.v, (short)0, acc[t], false, false);
        }
        __syncthreads();
    }

    for (int t = 0; t < 4; ++t)
        for (int v = 0; v < 8; ++v) {
            int r = row0 + wave * 16 + v + ((lane >> 4) << 3);
            int c = col0 + t * 16 + (lane & 15);
            Pb[(size_t)r * D + c] = (_Float16)acc[t][v];
        }
}

// ---------------------------------------------------------------------------
// Fused K/V projection: K[b] = yf[b] @ W2, V[b] = yf[b] @ W3.
// Shares the staged y-tile (the dominant HBM stream: 33.5 MB read once
// instead of twice) -> 8 WMMAs per A-fragment load.
// ---------------------------------------------------------------------------
__global__ __launch_bounds__(256) void projkv_kernel(
    const float* __restrict__ Y, const float* __restrict__ Wk,
    const float* __restrict__ Wv, _Float16* __restrict__ Kf,
    _Float16* __restrict__ Vf)
{
    const int D = 512, M = 1024;
    const int tm = blockIdx.x, tn = blockIdx.y, b = blockIdx.z;
    const int tid = threadIdx.x, lane = tid & 31, wave = tid >> 5;

    const float* Yb = Y + (size_t)b * M * D;   // yf = y.reshape(-1, ly, d)
    _Float16*    Kb = Kf + (size_t)b * M * D;
    _Float16*    Vb = Vf + (size_t)b * M * D;

    __shared__ alignas(16) _Float16 As[128 * PSTRIDE];
    __shared__ alignas(16) _Float16 Bks[64 * PSTRIDE];
    __shared__ alignas(16) _Float16 Bvs[64 * PSTRIDE];

    const int row0 = tm * 128, col0 = tn * 64;
    const int hb = (lane >> 4) << 2;

    v8f ak[4], av[4];
    for (int t = 0; t < 4; ++t)
        for (int v = 0; v < 8; ++v) { ak[t][v] = 0.0f; av[t][v] = 0.0f; }

    const unsigned* As32 = (const unsigned*)As;
    const unsigned* Bk32 = (const unsigned*)Bks;
    const unsigned* Bv32 = (const unsigned*)Bvs;

    for (int kb = 0; kb < D; kb += 32) {
        for (int i = tid; i < 128 * 32; i += 256) {
            int r = i >> 5, c = i & 31;
            As[r * PSTRIDE + c] = (_Float16)Yb[(size_t)(row0 + r) * D + kb + c];
        }
        for (int i = tid; i < 64 * 32; i += 256) {
            int n = i >> 5, c = i & 31;
            Bks[n * PSTRIDE + c] = (_Float16)Wk[(size_t)(kb + c) * D + col0 + n];
            Bvs[n * PSTRIDE + c] = (_Float16)Wv[(size_t)(kb + c) * D + col0 + n];
        }
        __syncthreads();

        AF a;
        const unsigned* ap = As32 + (wave * 16 + (lane & 15)) * (PSTRIDE / 2) + hb;
        for (int j = 0; j < 8; ++j) a.u[j] = ap[jd(j)];

        for (int t = 0; t < 4; ++t) {
            const int nb = (t * 16 + (lane & 15)) * (PSTRIDE / 2) + hb;
            AF bk, bv;
            const unsigned* kp = Bk32 + nb;
            const unsigned* vp = Bv32 + nb;
            for (int j = 0; j < 8; ++j) { bk.u[j] = kp[jd(j)]; bv.u[j] = vp[jd(j)]; }
            ak[t] = __builtin_amdgcn_wmma_f32_16x16x32_f16(
                false, a.v, false, bk.v, (short)0, ak[t], false, false);
            av[t] = __builtin_amdgcn_wmma_f32_16x16x32_f16(
                false, a.v, false, bv.v, (short)0, av[t], false, false);
        }
        __syncthreads();
    }

    for (int t = 0; t < 4; ++t)
        for (int v = 0; v < 8; ++v) {
            int r = row0 + wave * 16 + v + ((lane >> 4) << 3);
            int c = col0 + t * 16 + (lane & 15);
            Kb[(size_t)r * D + c] = (_Float16)ak[t][v];
            Vb[(size_t)r * D + c] = (_Float16)av[t][v];
        }
}

// ---------------------------------------------------------------------------
// Fused attention: per (batch, head, 128 q-rows). Flash-style streaming
// softmax over key blocks of 64 with exact-zero masking (padded keys give
// raw score == 0.0 exactly in any precision).
// ---------------------------------------------------------------------------
__global__ __launch_bounds__(256) void attn_kernel(
    const _Float16* __restrict__ Qf, const _Float16* __restrict__ Kf,
    const _Float16* __restrict__ Vf, _Float16* __restrict__ Of)
{
    const int D = 512, LX = 512, LY = 1024, HD = 64;
    const int qt = blockIdx.x;   // 0..3
    const int h  = blockIdx.y;   // 0..7
    const int b  = blockIdx.z;   // 0..15
    const int tid = threadIdx.x, lane = tid & 31, wave = tid >> 5;

    const _Float16* Qb = Qf + (size_t)b * LX * D;
    const _Float16* Kb = Kf + (size_t)b * LY * D;
    const _Float16* Vb = Vf + (size_t)b * LY * D;
    _Float16*       Ob = Of + (size_t)b * LX * D;

    __shared__ alignas(16) _Float16 Ks[64 * ASTRIDE];    // [key][hd]
    __shared__ alignas(16) _Float16 Vs[64 * ASTRIDE];    // [hd][key] (transposed)
    __shared__ alignas(16) _Float16 Ps[128 * ASTRIDE];   // [qrow][key]

    const int hb = (lane >> 4) << 2;

    // Resident Q A-fragments (hd = 64 -> two K=32 chunks); dword indexing.
    AF aq[2];
    {
        const int qrow = qt * 128 + wave * 16 + (lane & 15);
        const unsigned* qp = (const unsigned*)Qb + (size_t)qrow * (D / 2) + h * (HD / 2) + hb;
        for (int c = 0; c < 2; ++c)
            for (int j = 0; j < 8; ++j)
                aq[c].u[j] = qp[c * 16 + jd(j)];
    }

    v8f o[4];
    for (int t = 0; t < 4; ++t)
        for (int v = 0; v < 8; ++v) o[t][v] = 0.0f;
    float mrow[8], lrow[8];
    for (int v = 0; v < 8; ++v) { mrow[v] = -__builtin_inff(); lrow[v] = 0.0f; }
    const float scale = 0.125f;  // 1/sqrt(64)

    const unsigned* Ks32 = (const unsigned*)Ks;
    const unsigned* Vs32 = (const unsigned*)Vs;
    const unsigned* Ps32 = (const unsigned*)Ps;

    for (int kb = 0; kb < LY; kb += 64) {
#if defined(USE_ASYNC_LDS)
        // K tile: straight f16 copy -> async DMA to LDS, 16B per op,
        // tracked by ASYNCcnt (no VGPR round trip).
        for (int i = tid; i < 64 * 8; i += 256) {
            int r = i >> 3, c = (i & 7) * 8;  // c in halves, 8 halves = 16B
            const _Float16* gp = Kb + (size_t)(kb + r) * D + h * HD + c;
            _Float16*       lp = &Ks[r * ASTRIDE + c];
            __builtin_amdgcn_global_load_async_to_lds_b128(
                (g128_t)(uintptr_t)gp,
                (l128_t)(unsigned)(uintptr_t)lp,
                0, 0);
        }
        // V tile needs a transpose -> synchronous staging.
        for (int i = tid; i < 64 * 64; i += 256) {
            int r = i >> 6, c = i & 63;
            Vs[c * ASTRIDE + r] = Vb[(size_t)(kb + r) * D + h * HD + c];
        }
        __builtin_amdgcn_s_wait_asynccnt(0);
        __syncthreads();
#else
        for (int i = tid; i < 64 * 64; i += 256) {
            int r = i >> 6, c = i & 63;
            Ks[r * ASTRIDE + c] = Kb[(size_t)(kb + r) * D + h * HD + c];
            Vs[c * ASTRIDE + r] = Vb[(size_t)(kb + r) * D + h * HD + c];
        }
        __syncthreads();
#endif

        // S = Q @ K^T : 16 x 64 per wave (4 column tiles)
        v8f s[4];
        for (int t = 0; t < 4; ++t)
            for (int v = 0; v < 8; ++v) s[t][v] = 0.0f;
        for (int t = 0; t < 4; ++t) {
            const unsigned* kp = Ks32 + (t * 16 + (lane & 15)) * (ASTRIDE / 2) + hb;
            for (int c = 0; c < 2; ++c) {
                AF bk;
                for (int j = 0; j < 8; ++j) bk.u[j] = kp[c * 16 + jd(j)];
                s[t] = __builtin_amdgcn_wmma_f32_16x16x32_f16(
                    false, aq[c].v, false, bk.v, (short)0, s[t], false, false);
            }
        }

        // Streaming masked softmax. C-layout row = v + 8*(lane/16); each row
        // lives on a 16-lane half-group -> butterfly reductions stay in-half.
        for (int v = 0; v < 8; ++v) {
            float sv[4], sm = -__builtin_inff();
            for (int t = 0; t < 4; ++t) {
                sv[t] = s[t][v] * scale;           // masked entries are exactly 0
                sm = fmaxf(sm, sv[t]);             // reference max includes zeros
            }
            for (int off = 1; off < 16; off <<= 1)
                sm = fmaxf(sm, __shfl_xor(sm, off, 32));
            const float mnew  = fmaxf(mrow[v], sm);
            const float alpha = __expf(mrow[v] - mnew);   // exp(-inf)=0 first pass
            float rs = 0.0f, pv[4];
            for (int t = 0; t < 4; ++t) {
                float e = (s[t][v] == 0.0f) ? 0.0f : __expf(sv[t] - mnew);
                pv[t] = e; rs += e;
            }
            for (int off = 1; off < 16; off <<= 1)
                rs += __shfl_xor(rs, off, 32);
            lrow[v] = lrow[v] * alpha + rs;
            mrow[v] = mnew;
            for (int t = 0; t < 4; ++t) o[t][v] *= alpha;

            const int pr = wave * 16 + v + ((lane >> 4) << 3);
            for (int t = 0; t < 4; ++t)
                Ps[pr * ASTRIDE + t * 16 + (lane & 15)] = (_Float16)pv[t];
        }
        __syncthreads();

        // O += P @ V : contraction over 64 keys = two K=32 chunks
        for (int c = 0; c < 2; ++c) {
            AF ap;
            const unsigned* pp = Ps32 + (wave * 16 + (lane & 15)) * (ASTRIDE / 2) + hb;
            for (int j = 0; j < 8; ++j) ap.u[j] = pp[c * 16 + jd(j)];
            for (int t = 0; t < 4; ++t) {
                AF bv;
                const unsigned* vp = Vs32 + (t * 16 + (lane & 15)) * (ASTRIDE / 2) + hb;
                for (int j = 0; j < 8; ++j) bv.u[j] = vp[c * 16 + jd(j)];
                o[t] = __builtin_amdgcn_wmma_f32_16x16x32_f16(
                    false, ap.v, false, bv.v, (short)0, o[t], false, false);
            }
        }
        __syncthreads();
    }

    // Normalize by (sum + 1e-10) and store concat-head layout [LX][D] f16
    for (int v = 0; v < 8; ++v) {
        const float inv = 1.0f / (lrow[v] + 1e-10f);
        const int r = qt * 128 + wave * 16 + v + ((lane >> 4) << 3);
        for (int t = 0; t < 4; ++t) {
            const int c = h * HD + t * 16 + (lane & 15);
            Ob[(size_t)r * D + c] = (_Float16)(o[t][v] * inv);
        }
    }
}

// ---------------------------------------------------------------------------
// Output projection: out[b] = O[b] (512x512 f16) @ W4 (512x64 f32->f16),
// scattered to [nsx][nsy][lx][64] with b = sy*4 + sx.
// ---------------------------------------------------------------------------
__global__ __launch_bounds__(256) void outproj_kernel(
    const _Float16* __restrict__ Of, const float* __restrict__ W4,
    float* __restrict__ out)
{
    const int D = 512, LX = 512, HD = 64;
    const int tm = blockIdx.x;   // 0..3
    const int b  = blockIdx.y;   // 0..15
    const int tid = threadIdx.x, lane = tid & 31, wave = tid >> 5;

    const _Float16* Obf = Of + (size_t)b * LX * D;
    const int sy = b >> 2, sx = b & 3;
    float* outb = out + (size_t)(sx * 4 + sy) * LX * HD;

    __shared__ alignas(16) _Float16 Bs[64 * PSTRIDE];  // [n][k]

    const int row0 = tm * 128;
    const int hb = (lane >> 4) << 2;
    v8f acc[4];
    for (int t = 0; t < 4; ++t)
        for (int v = 0; v < 8; ++v) acc[t][v] = 0.0f;

    const unsigned* Bs32 = (const unsigned*)Bs;

    for (int kb = 0; kb < D; kb += 32) {
        for (int i = tid; i < 64 * 32; i += 256) {
            int n = i >> 5, c = i & 31;
            Bs[n * PSTRIDE + c] = (_Float16)W4[(size_t)(kb + c) * HD + n];
        }
        __syncthreads();

        AF a;
        const unsigned* op = (const unsigned*)Obf
            + (size_t)(row0 + wave * 16 + (lane & 15)) * (D / 2) + (kb >> 1) + hb;
        for (int j = 0; j < 8; ++j) a.u[j] = op[jd(j)];

        for (int t = 0; t < 4; ++t) {
            AF bf;
            const unsigned* bp = Bs32 + (t * 16 + (lane & 15)) * (PSTRIDE / 2) + hb;
            for (int j = 0; j < 8; ++j) bf.u[j] = bp[jd(j)];
            acc[t] = __builtin_amdgcn_wmma_f32_16x16x32_f16(
                false, a.v, false, bf.v, (short)0, acc[t], false, false);
        }
        __syncthreads();
    }

    for (int t = 0; t < 4; ++t)
        for (int v = 0; v < 8; ++v) {
            int r = row0 + wave * 16 + v + ((lane >> 4) << 3);
            int c = t * 16 + (lane & 15);
            outb[(size_t)r * HD + c] = acc[t][v];
        }
}

// ---------------------------------------------------------------------------
extern "C" void kernel_launch(void* const* d_in, const int* in_sizes, int n_in,
                              void* d_out, int out_size, void* d_ws, size_t ws_size,
                              hipStream_t stream)
{
    (void)in_sizes; (void)n_in; (void)out_size; (void)ws_size;
    const float* x  = (const float*)d_in[0];
    const float* y  = (const float*)d_in[1];
    const float* W1 = (const float*)d_in[2];
    const float* W2 = (const float*)d_in[3];
    const float* W3 = (const float*)d_in[4];
    const float* W4 = (const float*)d_in[5];
    float* out = (float*)d_out;

    // Workspace (f16): Q 8MB | K 16MB | V 16MB | O 8MB  (~48MB, L2-resident)
    _Float16* Qf = (_Float16*)d_ws;
    _Float16* Kf = Qf + (size_t)16 * 512  * 512;
    _Float16* Vf = Kf + (size_t)16 * 1024 * 512;
    _Float16* Of = Vf + (size_t)16 * 1024 * 512;

    dim3 blk(256);
    projq_kernel <<<dim3(4, 8, 16), blk, 0, stream>>>(x, W1, Qf);
    projkv_kernel<<<dim3(8, 8, 16), blk, 0, stream>>>(y, W2, W3, Kf, Vf);
    attn_kernel  <<<dim3(4, 8, 16), blk, 0, stream>>>(Qf, Kf, Vf, Of);
    outproj_kernel<<<dim3(4, 16),   blk, 0, stream>>>(Of, W4, out);
}
